// StaticNodeGNN_87479893885368
// MI455X (gfx1250) — matrix operand order repaired and checked by reading, not compiled, not run on hardware
//
#include <hip/hip_runtime.h>
#include <hip/hip_bf16.h>
#include <cstdint>
#include <cstddef>

// ---------------------------------------------------------------------------
// GNN pipeline for gfx1250 (MI455X): bf16 WMMA GEMMs with async-LDS B staging,
// pre-packed bf16 A-fragments, atomic edge scatter for GCN aggregation.
// ---------------------------------------------------------------------------

#define NN 50000      // nodes (divisible by 16)
#define NE 300000     // edges
#define HID 256       // hidden dim (all GEMM outputs are [N,256])
#define RB_CNT (NN / 16)                  // 3125 row-blocks
#define PACK_TOTAL (RB_CNT * 8 * 32)      // (rb, kb, lane) triples per tensor

typedef __attribute__((ext_vector_type(16))) __bf16          v16bf;
typedef __attribute__((ext_vector_type(8)))  float           v8f;

union Frag { uint4 q[2]; v16bf v; };      // 32B = one 16-lane-striped fragment

__device__ __forceinline__ unsigned short f2bf(float f) {
    unsigned int u = __float_as_uint(f);
    u += 0x7fffu + ((u >> 16) & 1u);      // round-to-nearest-even
    return (unsigned short)(u >> 16);
}

// ---------------------------------------------------------------------------
// Weight swizzle: fp32 [K,256] row-major -> bf16 B-fragment layout.
// ushort index = ((kb*16 + nt)*32 + lane)*16 + e
//   lane 0-15  : col = nt*16 + lane,      K = kb*32 + (e&7) + 16*(e>=8)
//   lane 16-31 : col = nt*16 + lane-16,   K = kb*32 + 8 + (e&7) + 16*(e>=8)
// ---------------------------------------------------------------------------
__global__ void swizzle_w_kernel(const float* __restrict__ W,
                                 unsigned short* __restrict__ Wz, int K) {
    int idx = blockIdx.x * blockDim.x + threadIdx.x;
    int total = (K >> 5) * 16 * 32;
    if (idx >= total) return;
    int lane = idx & 31;
    int nt   = (idx >> 5) & 15;
    int kb   = idx >> 9;
    int col  = nt * 16 + (lane & 15);
    int krow = kb * 32 + ((lane >> 4) << 3);
    unsigned short tmp[16];
#pragma unroll
    for (int e = 0; e < 16; ++e)
        tmp[e] = f2bf(W[(size_t)(krow + (e & 7) + ((e >> 3) << 4)) * HID + col]);
    uint4* dst = (uint4*)(Wz) + (size_t)idx * 2;
    dst[0] = ((const uint4*)tmp)[0];
    dst[1] = ((const uint4*)tmp)[1];
}

// ---------------------------------------------------------------------------
// Activation pack: fp32 [N,256] row-major (+ optional bias & relu) -> bf16
// A-fragment layout.  uint4 index = ((rb*8 + kb)*32 + lane)*2
//   element e: row = rb*16 + lane%16, col = kb*32 + 8*(lane/16)
//              + (e&7) + 16*(e>=8)   (two runs of 8 contiguous cols)
// ---------------------------------------------------------------------------
__global__ void pack_a_kernel(const float* __restrict__ src,
                              const float* __restrict__ bias, int act,
                              uint4* __restrict__ dst) {
    int idx = blockIdx.x * blockDim.x + threadIdx.x;
    if (idx >= PACK_TOTAL) return;
    int lane = idx & 31;
    int kb   = (idx >> 5) & 7;
    int rb   = idx >> 8;
    int row  = rb * 16 + (lane & 15);
    int c0   = kb * 32 + 8 * (lane >> 4);

    const float* p = src + (size_t)row * HID + c0;
    float v[16];
#pragma unroll
    for (int j = 0; j < 8; ++j) v[j]     = p[j];
#pragma unroll
    for (int j = 0; j < 8; ++j) v[8 + j] = p[16 + j];

    if (bias) {
        const float* bp = bias + c0;
#pragma unroll
        for (int j = 0; j < 8; ++j) v[j]     += bp[j];
#pragma unroll
        for (int j = 0; j < 8; ++j) v[8 + j] += bp[16 + j];
    }
    if (act) {
#pragma unroll
        for (int j = 0; j < 16; ++j) v[j] = fmaxf(v[j], 0.0f);
    }
    unsigned short tmp[16];
#pragma unroll
    for (int j = 0; j < 16; ++j) tmp[j] = f2bf(v[j]);
    dst[(size_t)idx * 2]     = ((const uint4*)tmp)[0];
    dst[(size_t)idx * 2 + 1] = ((const uint4*)tmp)[1];
}

// ---------------------------------------------------------------------------
// WMMA GEMM: Out[N,256] = act( A[N,K] @ W[K,256] + bias ), K = KB*32
//  - A pre-packed bf16 fragments; kb < 8 reads A0p, kb >= 8 reads A1p
//    (fused concat for the MLP GEMM; all packed tensors have 8 k-blocks)
//  - 256 threads = 8 waves; 128 rows x 128 cols per block; 8 C-tiles/wave
//  - B slab staged LDS-resident once per block via global_load_async_to_lds
//  - A ping-pong double buffered (unroll-by-2, no register rotation)
// Dynamic LDS: KB * 8KB  (64KB for KB=8, 128KB for KB=16)
// ---------------------------------------------------------------------------
__global__ __launch_bounds__(256) void gemm_wmma_kernel(
    const uint4* __restrict__ A0p, const uint4* __restrict__ A1p, int KB,
    const unsigned short* __restrict__ Wz, const float* __restrict__ bias,
    float* __restrict__ Out, int N, int act) {

    extern __shared__ char smem[];

    const int lane = threadIdx.x & 31;
    const int wave = threadIdx.x >> 5;
    const int half = lane >> 4;
    const int l16  = lane & 15;

    const int rowBase = blockIdx.x * 128 + wave * 16;
    const int colBase = blockIdx.y * 128;
    const int nt0     = colBase >> 4;          // first of 8 column tiles

    int rb = rowBase >> 4;
    if (rb >= RB_CNT) rb = RB_CNT - 1;         // clamp; OOB rows never stored

    // ---- async stage of the block's whole B slab into LDS ----------------
    {
        const unsigned ldsBase = (unsigned)(uintptr_t)smem;
        const int tid = threadIdx.x;
        for (int kb = 0; kb < KB; ++kb) {
#pragma unroll
            for (int p = 0; p < 2; ++p) {
                int idx = tid + p * 256;                       // 512 x 16B = 8KB
                unsigned loff = ldsBase + kb * 8192 + idx * 16;
                const char* g = (const char*)Wz +
                                ((size_t)(kb * 16 + nt0) << 10) + idx * 16;
                asm volatile("global_load_async_to_lds_b128 %0, %1, off"
                             :: "v"(loff), "v"(g) : "memory");
            }
        }
        asm volatile("s_wait_asynccnt 0x0" ::: "memory");
        __syncthreads();
    }

    v8f c[8] = {};
    const uint4* sB4 = (const uint4*)smem;

    auto loadA = [&](Frag& f, int kb) {
        const uint4* P = (kb < 8) ? A0p : A1p;
        int qi = (((rb << 3) + (kb & 7)) * 32 + lane) * 2;
        f.q[0] = P[qi];
        f.q[1] = P[qi + 1];
    };
    auto wmma8 = [&](const v16bf& a, int kb) {
        const int sbase = kb * 512 + lane * 2;      // uint4 units, 8KB/chunk
#pragma unroll
        for (int t = 0; t < 8; ++t) {
            Frag bf;
            bf.q[0] = sB4[sbase + t * 64];
            bf.q[1] = sB4[sbase + t * 64 + 1];
            c[t] = __builtin_amdgcn_wmma_f32_16x16x32_bf16(
                false, a, false, bf.v, (short)0, c[t], false, false);
        }
    };

    // ---- main K loop: ping-pong A fragments, 8 WMMAs per k-step ----------
    Frag a0, a1;
    loadA(a0, 0);
    loadA(a1, 1);
    for (int kb = 0; kb < KB; kb += 2) {
        wmma8(a0.v, kb);
        if (kb + 2 < KB) loadA(a0, kb + 2);
        wmma8(a1.v, kb + 1);
        if (kb + 3 < KB) loadA(a1, kb + 3);
    }

    // ---- store: C/D layout -> row = rowBase + r + 8*half, col = l16 ------
#pragma unroll
    for (int t = 0; t < 8; ++t) {
        const int col = colBase + t * 16 + l16;
        const float bv = bias ? bias[col] : 0.0f;
#pragma unroll
        for (int r = 0; r < 8; ++r) {
            const int orow = rowBase + r + 8 * half;
            if (orow < N) {
                float v = c[t][r] + bv;
                if (act) v = v > 0.0f ? v : 0.0f;
                Out[(size_t)orow * HID + col] = v;
            }
        }
    }
}

// ---------------------------------------------------------------------------
// GCN normalization + aggregation helpers
// ---------------------------------------------------------------------------
__global__ void deg_init_kernel(float* __restrict__ deg, int n) {
    int i = blockIdx.x * blockDim.x + threadIdx.x;
    if (i < n) deg[i] = 1.0f;                       // self loop
}

__global__ void deg_count_kernel(const long long* __restrict__ ei,
                                 float* __restrict__ deg, int ne) {
    int e = blockIdx.x * blockDim.x + threadIdx.x;
    if (e < ne) atomicAdd(&deg[(int)ei[NE + e]], 1.0f);   // col = target
}

__global__ void deg_rsqrt_kernel(float* __restrict__ deg, int n) {
    int i = blockIdx.x * blockDim.x + threadIdx.x;
    if (i < n) deg[i] = rsqrtf(deg[i]);             // deg >= 1 always
}

// AGG[i,:] = dinv[i]^2 * T[i,:]   (self-loop term initializes accumulator)
__global__ void agg_init_kernel(const float* __restrict__ T,
                                const float* __restrict__ dinv,
                                float* __restrict__ AGG, int n4) {
    int i = blockIdx.x * blockDim.x + threadIdx.x;
    if (i >= n4) return;
    float d = dinv[i >> 6];                         // 64 float4 per node
    float s = d * d;
    float4 t = ((const float4*)T)[i];
    ((float4*)AGG)[i] = make_float4(t.x * s, t.y * s, t.z * s, t.w * s);
}

// One wave per edge: AGG[dst,:] += dinv[src]*dinv[dst] * T[src,:]
__global__ __launch_bounds__(256) void agg_edges_kernel(
    const long long* __restrict__ ei, const float* __restrict__ T,
    const float* __restrict__ dinv, float* __restrict__ AGG) {
    int gid  = blockIdx.x * blockDim.x + threadIdx.x;
    int e    = gid >> 5;
    int lane = gid & 31;
    if (e >= NE) return;
    int src = (int)ei[e];
    int dst = (int)ei[NE + e];
    float s = dinv[src] * dinv[dst];
    const float* tp = T + (size_t)src * HID;
    float*       ap = AGG + (size_t)dst * HID;
#pragma unroll
    for (int j = 0; j < 8; ++j) {
        int f = lane + j * 32;
        atomicAdd(ap + f, tp[f] * s);
    }
}

// logits[i] = dot(hidden[i,:], Wm2) + bm2  — one wave per node
__global__ __launch_bounds__(256) void mlp2_kernel(
    const float* __restrict__ Hid, const float* __restrict__ Wm2,
    const float* __restrict__ bm2, float* __restrict__ out) {
    int gid  = blockIdx.x * blockDim.x + threadIdx.x;
    int node = gid >> 5;
    int lane = gid & 31;
    if (node >= NN) return;
    float acc = 0.0f;
#pragma unroll
    for (int j = 0; j < 8; ++j) {
        int f = lane + j * 32;
        acc += Hid[(size_t)node * HID + f] * Wm2[f];
    }
#pragma unroll
    for (int o = 16; o > 0; o >>= 1) acc += __shfl_xor(acc, o, 32);
    if (lane == 0) out[node] = acc + bm2[0];
}

// ---------------------------------------------------------------------------
// Host launcher
// ---------------------------------------------------------------------------
extern "C" void kernel_launch(void* const* d_in, const int* in_sizes, int n_in,
                              void* d_out, int out_size, void* d_ws, size_t ws_size,
                              hipStream_t stream) {
    const float*     x    = (const float*)d_in[0];
    const long long* ei   = (const long long*)d_in[1];
    const float*     W_in = (const float*)d_in[2];
    const float*     b_in = (const float*)d_in[3];
    const float*     W1   = (const float*)d_in[4];
    const float*     b1   = (const float*)d_in[5];
    const float*     W2   = (const float*)d_in[6];
    const float*     b2   = (const float*)d_in[7];
    const float*     Wm1  = (const float*)d_in[8];
    const float*     bm1  = (const float*)d_in[9];
    const float*     Wm2  = (const float*)d_in[10];
    const float*     bm2  = (const float*)d_in[11];
    float* logits = (float*)d_out;

    // Workspace carve-out (256B aligned)
    size_t off = 0;
    auto carve = [&](size_t bytes) -> void* {
        void* p = (char*)d_ws + off;
        off += (bytes + 255) & ~(size_t)255;
        return p;
    };
    unsigned short* WzIn = (unsigned short*)carve(256 * 256 * 2);
    unsigned short* Wz1  = (unsigned short*)carve(256 * 256 * 2);
    unsigned short* Wz2  = (unsigned short*)carve(256 * 256 * 2);
    unsigned short* WzM1 = (unsigned short*)carve(512 * 256 * 2);
    float* dinv = (float*)carve((size_t)NN * 4);
    uint4* Xp   = (uint4*)carve((size_t)PACK_TOTAL * 32);   // packed x
    uint4* HINp = (uint4*)carve((size_t)PACK_TOTAL * 32);   // packed h_in
    uint4* Hp   = (uint4*)carve((size_t)PACK_TOTAL * 32);   // packed h1 / h2
    float* HIN  = (float*)carve((size_t)NN * HID * 4);      // h_in fp32 (transient)
    float* T    = (float*)carve((size_t)NN * HID * 4);      // h@W
    float* AGG  = (float*)carve((size_t)NN * HID * 4);      // aggregation accum
    float* HIDDEN = T;                                      // T dead by MLP time

    const int n4 = NN * (HID / 4);
    const int packBlocks = PACK_TOTAL / 256;                // 3125 exactly

    // 1) Swizzle weights into WMMA B-fragment layout
    swizzle_w_kernel<<<(256/32)*512/256, 256, 0, stream>>>(W_in, WzIn, 256);
    swizzle_w_kernel<<<(256/32)*512/256, 256, 0, stream>>>(W1,   Wz1,  256);
    swizzle_w_kernel<<<(256/32)*512/256, 256, 0, stream>>>(W2,   Wz2,  256);
    swizzle_w_kernel<<<(512/32)*512/256, 256, 0, stream>>>(Wm1,  WzM1, 512);

    // 2) Degree -> dinv (shared by both convolutions)
    deg_init_kernel<<<(NN + 255) / 256, 256, 0, stream>>>(dinv, NN);
    deg_count_kernel<<<(NE + 255) / 256, 256, 0, stream>>>(ei, dinv, NE);
    deg_rsqrt_kernel<<<(NN + 255) / 256, 256, 0, stream>>>(dinv, NN);

    dim3 ggrid((NN + 127) / 128, HID / 128);
    const size_t lds8  = 8  * 8192;   // 64KB
    const size_t lds16 = 16 * 8192;   // 128KB

    // 3) h_in = relu(x @ W_in + b_in)   (x packed to bf16 fragments first)
    pack_a_kernel<<<packBlocks, 256, 0, stream>>>(x, nullptr, 0, Xp);
    gemm_wmma_kernel<<<ggrid, 256, lds8, stream>>>(Xp, Xp, 8, WzIn, b_in, HIN, NN, 1);
    pack_a_kernel<<<packBlocks, 256, 0, stream>>>(HIN, nullptr, 0, HINp);

    // 4) conv1: T = h_in @ W1 ; AGG = norm-aggregate ; Hp = pack(relu(AGG+b1))
    gemm_wmma_kernel<<<ggrid, 256, lds8, stream>>>(HINp, HINp, 8, Wz1, nullptr, T, NN, 0);
    agg_init_kernel<<<(n4 + 255) / 256, 256, 0, stream>>>(T, dinv, AGG, n4);
    agg_edges_kernel<<<((size_t)NE * 32 + 255) / 256, 256, 0, stream>>>(ei, T, dinv, AGG);
    pack_a_kernel<<<packBlocks, 256, 0, stream>>>(AGG, b1, 1, Hp);

    // 5) conv2 (same pattern, reusing buffers)
    gemm_wmma_kernel<<<ggrid, 256, lds8, stream>>>(Hp, Hp, 8, Wz2, nullptr, T, NN, 0);
    agg_init_kernel<<<(n4 + 255) / 256, 256, 0, stream>>>(T, dinv, AGG, n4);
    agg_edges_kernel<<<((size_t)NE * 32 + 255) / 256, 256, 0, stream>>>(ei, T, dinv, AGG);
    pack_a_kernel<<<packBlocks, 256, 0, stream>>>(AGG, b2, 1, Hp);

    // 6) hidden = relu(concat(h_in, h2) @ Wm1 + bm1)  (split-KB, no concat copy)
    gemm_wmma_kernel<<<ggrid, 256, lds16, stream>>>(HINp, Hp, 16, WzM1, bm1, HIDDEN, NN, 1);

    // 7) logits = hidden @ Wm2 + bm2
    mlp2_kernel<<<((size_t)NN * 32 + 255) / 256, 256, 0, stream>>>(HIDDEN, Wm2, bm2, logits);
}
